// CausalSelfAttention_8899172237942
// MI455X (gfx1250) — matrix operand order repaired and checked by reading, not compiled, not run on hardware
//
#include <hip/hip_runtime.h>
#include <hip/hip_bf16.h>

typedef __bf16 bf16_t;
typedef __attribute__((ext_vector_type(16))) __bf16 v16bf;
typedef __attribute__((ext_vector_type(8)))  float  v8f;
typedef __attribute__((ext_vector_type(4)))  unsigned tdm_g0_t;
typedef __attribute__((ext_vector_type(8)))  unsigned tdm_g1_t;

union Frag {
    v16bf v;
    uint4 q[2];
};

#define DIMD 1024
#define SEQ  2048
#define NB   4
#define NH   16
#define HD   64

// ---------------------------------------------------------------------------
// CDNA5 data-mover helpers (gfx1250)
// ---------------------------------------------------------------------------
typedef __attribute__((address_space(3))) char lds_char_t;

__device__ __forceinline__ unsigned lds_off_u32(const void* p) {
    return (unsigned)(unsigned long long)(lds_char_t*)p;   // generic->LDS, 32-bit offset
}

// ASYNCcnt path: mem addr = SGPR base + 32-bit per-lane VGPR offset (GVS)
__device__ __forceinline__ void async_copy_b128(const void* lds_dst,
                                                unsigned gvoff,
                                                unsigned long long sbase) {
    asm volatile("global_load_async_to_lds_b128 %0, %1, %2"
                 :: "v"(lds_off_u32(lds_dst)), "v"(gvoff), "s"(sbase)
                 : "memory");
}

__device__ __forceinline__ void wait_asynccnt0() {
    asm volatile("s_wait_asynccnt 0x0" ::: "memory");
}

// TENSORcnt path: 2D bf16 tile load, 64 cols x 64 rows, dim0_stride=64 elems,
// LDS padding 4 DWORDs every 32 DWORDs -> LDS row pitch 144B (stride-72 bf16).
__device__ __forceinline__ void tdm_load_k_tile(unsigned lds_byte_off,
                                                unsigned long long gaddr) {
    tdm_g0_t g0;
    g0.x = 1u;                                    // count=1 (valid user D#)
    g0.y = lds_byte_off;                          // lds_addr
    g0.z = (unsigned)gaddr;                       // global_addr[31:0]
    g0.w = (unsigned)(gaddr >> 32) | 0x80000000u; // global_addr[56:32] | type=2
    tdm_g1_t g1 = {
        0x07110000u,            // data_size=2B | pad_en | pad_interval=32dw | pad_amt=4dw
        (unsigned)(HD) << 16,   // tensor_dim0 = 64
        (unsigned)(SEQ) << 16,  // tensor_dim1 = 2048
        (unsigned)(HD) << 16,   // tile_dim0 = 64
        (unsigned)(HD),         // tile_dim1 = 64, tile_dim2 = 0
        (unsigned)(HD),         // tensor_dim0_stride = 64
        0u, 0u                  // dim1_stride unused (2D)
    };
    asm volatile("tensor_load_to_lds %0, %1"
                 :: "s"(g0), "s"(g1)
                 : "memory");
}

// ---------------------------------------------------------------------------
// GEMM: C[M,N] = A[M,1024] * W[1024,N] + bias     (double-buffered LDS)
//  EPI 0: scatter bf16 into Q/K/V [B,H,S,Hd];  EPI 1: f32 out [M,N]
// ---------------------------------------------------------------------------
template<int EPI, bool A_BF16>
__global__ __launch_bounds__(256) void gemm_wmma_bf16(
    const void* __restrict__ Aptr, const float* __restrict__ W,
    const float* __restrict__ bias, float* __restrict__ outF,
    bf16_t* __restrict__ Qb, bf16_t* __restrict__ Kb, bf16_t* __restrict__ Vb,
    int N)
{
    constexpr int K = DIMD;
    const int m0   = blockIdx.y * 128;
    const int n0   = blockIdx.x * 128;
    const int tid  = threadIdx.x;
    const int lane = tid & 31;
    const int wave = tid >> 5;
    const int l16  = lane & 15;
    const int hsel = lane >> 4;
    const int wm   = wave >> 1;
    const int wn   = wave & 1;

    __shared__ bf16_t As[2][128 * 40];
    __shared__ bf16_t Bs[2][128 * 40];

    float4 aF[4];
    uint4  aH[2];
    float4 bF[4];

    auto gload = [&](int kt) {
        if constexpr (A_BF16) {
            const bf16_t* A = (const bf16_t*)Aptr;
#pragma unroll
            for (int i = 0; i < 2; ++i) {
                int c = tid + i * 256, row = c >> 2, kq = c & 3;
                aH[i] = *(const uint4*)&A[(size_t)(m0 + row) * K + kt + kq * 8];
            }
        } else {
            const float* A = (const float*)Aptr;
#pragma unroll
            for (int i = 0; i < 4; ++i) {
                int c = tid + i * 256, row = c >> 3, kq = c & 7;
                aF[i] = *(const float4*)&A[(size_t)(m0 + row) * K + kt + kq * 4];
            }
        }
#pragma unroll
        for (int i = 0; i < 4; ++i) {
            int c = tid + i * 256, k = c >> 5, nq = c & 31;
            bF[i] = *(const float4*)&W[(size_t)(kt + k) * N + n0 + nq * 4];
        }
    };
    auto lstore = [&](int buf) {
        if constexpr (A_BF16) {
#pragma unroll
            for (int i = 0; i < 2; ++i) {
                int c = tid + i * 256, row = c >> 2, kq = c & 3;
                *(uint4*)&As[buf][row * 40 + kq * 8] = aH[i];
            }
        } else {
#pragma unroll
            for (int i = 0; i < 4; ++i) {
                int c = tid + i * 256, row = c >> 3, kq = c & 7;
                bf16_t* d = &As[buf][row * 40 + kq * 4];
                d[0] = (bf16_t)aF[i].x; d[1] = (bf16_t)aF[i].y;
                d[2] = (bf16_t)aF[i].z; d[3] = (bf16_t)aF[i].w;
            }
        }
#pragma unroll
        for (int i = 0; i < 4; ++i) {
            int c = tid + i * 256, k = c >> 5, nq = c & 31;
            Bs[buf][(nq * 4 + 0) * 40 + k] = (bf16_t)bF[i].x;
            Bs[buf][(nq * 4 + 1) * 40 + k] = (bf16_t)bF[i].y;
            Bs[buf][(nq * 4 + 2) * 40 + k] = (bf16_t)bF[i].z;
            Bs[buf][(nq * 4 + 3) * 40 + k] = (bf16_t)bF[i].w;
        }
    };

    const v8f vzero = {0.f,0.f,0.f,0.f,0.f,0.f,0.f,0.f};
    v8f acc[2][4];
#pragma unroll
    for (int mi = 0; mi < 2; ++mi)
#pragma unroll
        for (int ni = 0; ni < 4; ++ni) acc[mi][ni] = vzero;

    gload(0);
    lstore(0);
    __syncthreads();

    int buf = 0;
    for (int kt = 0; kt < K; kt += 32) {
        const bool more = (kt + 32) < K;
        if (more) gload(kt + 32);
        if (kt + 64 < K) {
            if constexpr (!A_BF16) {
                const float* A = (const float*)Aptr;
                __builtin_prefetch(&A[(size_t)(m0 + (tid >> 3)) * K + kt + 64], 0, 1);
            }
            __builtin_prefetch(&W[(size_t)(kt + 64 + (tid >> 5)) * N + n0 + (tid & 31) * 4], 0, 1);
        }

        Frag af[2];
#pragma unroll
        for (int mi = 0; mi < 2; ++mi) {
            int row = wm * 32 + mi * 16 + l16;
            af[mi].q[0] = *(const uint4*)&As[buf][row * 40 + hsel * 8];
            af[mi].q[1] = *(const uint4*)&As[buf][row * 40 + 16 + hsel * 8];
        }
        Frag bfr[4];
#pragma unroll
        for (int ni = 0; ni < 4; ++ni) {
            int col = wn * 64 + ni * 16 + l16;
            bfr[ni].q[0] = *(const uint4*)&Bs[buf][col * 40 + hsel * 8];
            bfr[ni].q[1] = *(const uint4*)&Bs[buf][col * 40 + 16 + hsel * 8];
        }
#pragma unroll
        for (int mi = 0; mi < 2; ++mi)
#pragma unroll
            for (int ni = 0; ni < 4; ++ni)
                acc[mi][ni] = __builtin_amdgcn_wmma_f32_16x16x32_bf16(
                    false, af[mi].v, false, bfr[ni].v, (short)0, acc[mi][ni],
                    false, false);

        if (more) lstore(buf ^ 1);
        __syncthreads();
        buf ^= 1;
    }

#pragma unroll
    for (int ni = 0; ni < 4; ++ni) {
        int ng = n0 + wn * 64 + ni * 16 + l16;
        float bv = bias[ng];
        if constexpr (EPI == 0) {
            int which = ng >> 10;
            int dd = ng & 1023;
            int h = dd >> 6, hd = dd & 63;
            bf16_t* dst = (which == 0) ? Qb : (which == 1) ? Kb : Vb;
#pragma unroll
            for (int mi = 0; mi < 2; ++mi)
#pragma unroll
                for (int j = 0; j < 8; ++j) {
                    int m = m0 + wm * 32 + mi * 16 + j + hsel * 8;
                    int bb = m >> 11, s = m & 2047;
                    dst[(((size_t)(bb * NH + h)) * SEQ + s) * HD + hd] =
                        (bf16_t)(acc[mi][ni][j] + bv);
                }
        } else {
#pragma unroll
            for (int mi = 0; mi < 2; ++mi)
#pragma unroll
                for (int j = 0; j < 8; ++j) {
                    int m = m0 + wm * 32 + mi * 16 + j + hsel * 8;
                    outF[(size_t)m * N + ng] = acc[mi][ni][j] + bv;
                }
        }
    }
}

// ---------------------------------------------------------------------------
// Flash attention, double-buffered KV:
//   Q tile  : GLOBAL_LOAD_ASYNC_TO_LDS_B128   (ASYNCcnt)
//   K tiles : TENSOR_LOAD_TO_LDS, hw-padded to stride-72 rows  (TENSORcnt)
//   V tiles : register-staged manual transpose
// One barrier per KV iteration; tile t+1 streams while tile t computes.
// ---------------------------------------------------------------------------
__global__ __launch_bounds__(256) void attn_flash_wmma(
    const bf16_t* __restrict__ Qb, const bf16_t* __restrict__ Kb,
    const bf16_t* __restrict__ Vb, bf16_t* __restrict__ Ob)
{
    const int qt = blockIdx.x, h = blockIdx.y, b = blockIdx.z;
    const int qbase = qt * 128;
    const int tid  = threadIdx.x;
    const int lane = tid & 31;
    const int wv   = tid >> 5;
    const int l16  = lane & 15;
    const int hsel = lane >> 4;
    const size_t headoff = ((size_t)(b * NH + h)) * SEQ * HD;

    __shared__ bf16_t Qs[128 * 72];        // 18 KB   [q][d]
    __shared__ bf16_t Ks[2][64 * 72];      // 18 KB   [key][d]  (TDM-padded rows)
    __shared__ bf16_t Vts[2][64 * 72];     // 18 KB   [d][key]
    __shared__ bf16_t Ps[8][16 * 72];      // 18 KB   per-wave P scratch

    // ---- Q tile: async straight copy ----
    {
        const unsigned long long qsrc =
            (unsigned long long)(const void*)(Qb + headoff + (size_t)qbase * HD);
#pragma unroll
        for (int i = 0; i < 4; ++i) {
            int c = tid + i * 256, row = c >> 3, dq = c & 7;
            async_copy_b128(&Qs[row * 72 + dq * 8],
                            (unsigned)((row * HD + dq * 8) * 2), qsrc);
        }
    }

    // ---- V staging helpers (register double-buffer) ----
    uint4 vr[2];
    auto vload = [&](int kb) {
#pragma unroll
        for (int i = 0; i < 2; ++i) {
            int c = tid + i * 256, row = c >> 3, dq = c & 7;
            vr[i] = *(const uint4*)&Vb[headoff + (size_t)(kb + row) * HD + dq * 8];
        }
    };
    auto vstore = [&](int buf) {
#pragma unroll
        for (int i = 0; i < 2; ++i) {
            int c = tid + i * 256, row = c >> 3, dq = c & 7;
            union { uint4 q; bf16_t e[8]; } vv;
            vv.q = vr[i];
#pragma unroll
            for (int j = 0; j < 8; ++j)
                Vts[buf][(dq * 8 + j) * 72 + row] = vv.e[j];
        }
    };

    const int nkv = (qbase + 128) / 64;

    // ---- prologue: stage tile 0 into buffer 0 ----
    if (wv == 0)
        tdm_load_k_tile(lds_off_u32(&Ks[0][0]),
                        (unsigned long long)(const void*)(Kb + headoff));
    vload(0);
    vstore(0);
    wait_asynccnt0();                       // Q copies done (per wave)
    if (wv == 0) __builtin_amdgcn_s_wait_tensorcnt(0);
    __syncthreads();

    Frag qf[2];
    {
        int row = wv * 16 + l16;
#pragma unroll
        for (int ks = 0; ks < 2; ++ks) {
            qf[ks].q[0] = *(const uint4*)&Qs[row * 72 + ks * 32 + hsel * 8];
            qf[ks].q[1] = *(const uint4*)&Qs[row * 72 + ks * 32 + 16 + hsel * 8];
        }
    }

    const v8f vzero = {0.f,0.f,0.f,0.f,0.f,0.f,0.f,0.f};
    float mrow[8], lrow[8];
    v8f oacc[4];
#pragma unroll
    for (int j = 0; j < 8; ++j) { mrow[j] = -3.0e38f; lrow[j] = 0.f; }
#pragma unroll
    for (int ni = 0; ni < 4; ++ni) oacc[ni] = vzero;

    int buf = 0;
    for (int t = 0; t < nkv; ++t) {
        const int kb = t * 64;
        const bool more = (t + 1) < nkv;
        if (more) {
            if (wv == 0)
                tdm_load_k_tile(lds_off_u32(&Ks[buf ^ 1][0]),
                                (unsigned long long)(const void*)
                                    (Kb + headoff + (size_t)(kb + 64) * HD));
            vload(kb + 64);                 // global V loads in flight during compute
        }

        // ---- S = Q K^T ----
        v8f sacc[4];
#pragma unroll
        for (int ni = 0; ni < 4; ++ni) sacc[ni] = vzero;
#pragma unroll
        for (int ni = 0; ni < 4; ++ni) {
            int key = ni * 16 + l16;
#pragma unroll
            for (int ks = 0; ks < 2; ++ks) {
                Frag kf;
                kf.q[0] = *(const uint4*)&Ks[buf][key * 72 + ks * 32 + hsel * 8];
                kf.q[1] = *(const uint4*)&Ks[buf][key * 72 + ks * 32 + 16 + hsel * 8];
                sacc[ni] = __builtin_amdgcn_wmma_f32_16x16x32_bf16(
                    false, qf[ks].v, false, kf.v, (short)0, sacc[ni], false, false);
            }
        }

        // ---- causal mask + online softmax ----
        float mnew[8], alpha[8], psum[8];
#pragma unroll
        for (int j = 0; j < 8; ++j) { mnew[j] = mrow[j]; psum[j] = 0.f; }
#pragma unroll
        for (int ni = 0; ni < 4; ++ni) {
            int key = kb + ni * 16 + l16;
#pragma unroll
            for (int j = 0; j < 8; ++j) {
                int qi = qbase + wv * 16 + j + hsel * 8;
                float s = (key <= qi) ? sacc[ni][j] * 0.125f : -3.0e38f;
                sacc[ni][j] = s;
                mnew[j] = fmaxf(mnew[j], s);
            }
        }
#pragma unroll
        for (int j = 0; j < 8; ++j) {
#pragma unroll
            for (int off = 8; off; off >>= 1)
                mnew[j] = fmaxf(mnew[j], __shfl_xor(mnew[j], off, 32));
            alpha[j] = __expf(mrow[j] - mnew[j]);
        }
#pragma unroll
        for (int ni = 0; ni < 4; ++ni)
#pragma unroll
            for (int j = 0; j < 8; ++j) {
                float p = __expf(sacc[ni][j] - mnew[j]);
                sacc[ni][j] = p;
                psum[j] += p;
            }
#pragma unroll
        for (int j = 0; j < 8; ++j) {
#pragma unroll
            for (int off = 8; off; off >>= 1)
                psum[j] += __shfl_xor(psum[j], off, 32);
            lrow[j] = lrow[j] * alpha[j] + psum[j];
            mrow[j] = mnew[j];
        }
#pragma unroll
        for (int ni = 0; ni < 4; ++ni)
#pragma unroll
            for (int j = 0; j < 8; ++j) oacc[ni][j] *= alpha[j];

        // ---- P -> per-wave LDS (A-operand layout) ----
#pragma unroll
        for (int ni = 0; ni < 4; ++ni)
#pragma unroll
            for (int j = 0; j < 8; ++j)
                Ps[wv][(j + hsel * 8) * 72 + ni * 16 + l16] = (bf16_t)sacc[ni][j];

        // ---- O += P * V ---- (same-wave LDS ops in-order)
        Frag pf[2];
        {
            int row = l16;
#pragma unroll
            for (int ks = 0; ks < 2; ++ks) {
                pf[ks].q[0] = *(const uint4*)&Ps[wv][row * 72 + ks * 32 + hsel * 8];
                pf[ks].q[1] = *(const uint4*)&Ps[wv][row * 72 + ks * 32 + 16 + hsel * 8];
            }
        }
#pragma unroll
        for (int ni = 0; ni < 4; ++ni) {
            int d = ni * 16 + l16;
#pragma unroll
            for (int ks = 0; ks < 2; ++ks) {
                Frag vf;
                vf.q[0] = *(const uint4*)&Vts[buf][d * 72 + ks * 32 + hsel * 8];
                vf.q[1] = *(const uint4*)&Vts[buf][d * 72 + ks * 32 + 16 + hsel * 8];
                oacc[ni] = __builtin_amdgcn_wmma_f32_16x16x32_bf16(
                    false, pf[ks].v, false, vf.v, (short)0, oacc[ni], false, false);
            }
        }

        if (more) vstore(buf ^ 1);          // drain staged V regs to other buffer
        if (wv == 0) __builtin_amdgcn_s_wait_tensorcnt(0);
        __syncthreads();
        buf ^= 1;
    }

    // ---- normalize, emit bf16 [B,S,D] ----
#pragma unroll
    for (int ni = 0; ni < 4; ++ni)
#pragma unroll
        for (int j = 0; j < 8; ++j) {
            int s = qbase + wv * 16 + j + hsel * 8;
            int col = h * HD + ni * 16 + l16;
            Ob[((size_t)b * SEQ + s) * DIMD + col] = (bf16_t)(oacc[ni][j] / lrow[j]);
        }
}

// ---------------------------------------------------------------------------
extern "C" void kernel_launch(void* const* d_in, const int* in_sizes, int n_in,
                              void* d_out, int out_size, void* d_ws, size_t ws_size,
                              hipStream_t stream)
{
    (void)in_sizes; (void)n_in; (void)out_size; (void)ws_size;
    const float* x      = (const float*)d_in[0];
    const float* W_attn = (const float*)d_in[1];
    const float* b_attn = (const float*)d_in[2];
    const float* W_proj = (const float*)d_in[3];
    const float* b_proj = (const float*)d_in[4];
    float* out = (float*)d_out;

    const size_t HE = (size_t)NB * NH * SEQ * HD;  // 8,388,608 elements
    bf16_t* Qb = (bf16_t*)d_ws;
    bf16_t* Kb = Qb + HE;
    bf16_t* Vb = Kb + HE;
    bf16_t* Ab = Vb + HE;

    gemm_wmma_bf16<0, false><<<dim3(3 * DIMD / 128, NB * SEQ / 128), 256, 0, stream>>>(
        x, W_attn, b_attn, nullptr, Qb, Kb, Vb, 3 * DIMD);

    attn_flash_wmma<<<dim3(SEQ / 128, NH, NB), 256, 0, stream>>>(Qb, Kb, Vb, Ab);

    gemm_wmma_bf16<1, true><<<dim3(DIMD / 128, NB * SEQ / 128), 256, 0, stream>>>(
        Ab, W_proj, b_proj, out, nullptr, nullptr, nullptr, DIMD);
}
